// NeighborhoodMA1D_41678362640880
// MI455X (gfx1250) — matrix operand order repaired and checked by reading, not compiled, not run on hardware
//
#include <hip/hip_runtime.h>
#include <hip/hip_bf16.h>

typedef __attribute__((ext_vector_type(16))) _Float16 v16h;
typedef __attribute__((ext_vector_type(8)))  _Float16 v8h;
typedef __attribute__((ext_vector_type(8)))  float    v8f;

#define BN  2
#define LL  4096
#define CC  512
#define HH  8
#define KW  13
#define DHD 64

#if defined(__has_builtin)
#if __has_builtin(__builtin_amdgcn_sched_barrier)
#define SCHED_FENCE() __builtin_amdgcn_sched_barrier(0)
#endif
#endif
#ifndef SCHED_FENCE
#define SCHED_FENCE()
#endif

// ---------------------------------------------------------------------------
// fp32 -> f16 elementwise convert (activations), memory bound
// ---------------------------------------------------------------------------
__global__ __launch_bounds__(256) void cvt_f32_f16(const float* __restrict__ in,
                                                   _Float16* __restrict__ out, int n) {
  int i = blockIdx.x * 256 + threadIdx.x;
  if (i < n) out[i] = (_Float16)in[i];
}

// ---------------------------------------------------------------------------
// fp32 -> f16 transpose: Wt[n*512 + k] = W[k*512 + n]  (512x512)
// Makes B-fragments (16 consecutive K per lane, fixed column) contiguous.
// ---------------------------------------------------------------------------
__global__ __launch_bounds__(256) void cvt_transpose_f16(const float* __restrict__ W,
                                                         _Float16* __restrict__ Wt) {
  int i = blockIdx.x * 256 + threadIdx.x;   // i = n*512 + k
  int n = i >> 9;
  int k = i & 511;
  Wt[i] = (_Float16)W[k * CC + n];
}

// ---------------------------------------------------------------------------
// GEMM fragments for one 32-deep K step: two 16x32 A tiles + four 32x16 B tiles
// ---------------------------------------------------------------------------
struct Frags {
  v8h a0lo, a0hi, a1lo, a1hi;
  v16h b0, b1, b2, b3;
};

union AV { v16h v; v8h h[2]; };

// ---------------------------------------------------------------------------
// C[m,n] = sum_k A[m,k]*Bt[n,k] + bias[n]
// A: M x 512 f16 row-major; Bt: 512 x 512 f16 (transposed weight); C fp32.
// Block: 256 threads = 8 waves. Wave w -> 32(M) x 64(N) tile, software-
// pipelined k-loop (load k+1 fragments while WMMAs for k execute), with
// sched_barrier fences so the scheduler cannot re-serialize load->wait->wmma.
// grid.x = M/32.
// ---------------------------------------------------------------------------
__global__ __launch_bounds__(256) void gemm_wmma_f16(const _Float16* __restrict__ A,
                                                     const _Float16* __restrict__ Bt,
                                                     const float* __restrict__ bias,
                                                     float* __restrict__ Cout) {
  const int lane = threadIdx.x & 31;
  const int wave = threadIdx.x >> 5;
  const int m0   = blockIdx.x * 32;
  const int n0   = wave * 64;
  const int l16  = lane & 15;
  const int kSel = lane >> 4;     // 0: low-K half, 1: high-K half (contiguous k = 16*kSel + s, same for A and B)

  const _Float16* arow0 = A + (size_t)(m0 + l16) * CC;
  const _Float16* arow1 = A + (size_t)(m0 + 16 + l16) * CC;
  const _Float16* bcol  = Bt + (size_t)(n0 + l16) * CC + kSel * 16;

  v8f acc[8] = {};

  // fragment loader for K step starting at kk
  auto load_frags = [&](int kk) {
    Frags f;
    f.a0lo = *(const v8h*)(arow0 + kk + kSel * 16);
    f.a0hi = *(const v8h*)(arow0 + kk + 8 + kSel * 16);
    f.a1lo = *(const v8h*)(arow1 + kk + kSel * 16);
    f.a1hi = *(const v8h*)(arow1 + kk + 8 + kSel * 16);
    f.b0 = *(const v16h*)(bcol + kk);
    f.b1 = *(const v16h*)(bcol + 16 * CC + kk);
    f.b2 = *(const v16h*)(bcol + 32 * CC + kk);
    f.b3 = *(const v16h*)(bcol + 48 * CC + kk);
    return f;
  };

  auto compute = [&](const Frags& f) {
    AV a0, a1;
    a0.h[0] = f.a0lo; a0.h[1] = f.a0hi;
    a1.h[0] = f.a1lo; a1.h[1] = f.a1hi;
    acc[0] = __builtin_amdgcn_wmma_f32_16x16x32_f16(false, a0.v, false, f.b0, (short)0, acc[0], false, false);
    acc[1] = __builtin_amdgcn_wmma_f32_16x16x32_f16(false, a0.v, false, f.b1, (short)0, acc[1], false, false);
    acc[2] = __builtin_amdgcn_wmma_f32_16x16x32_f16(false, a0.v, false, f.b2, (short)0, acc[2], false, false);
    acc[3] = __builtin_amdgcn_wmma_f32_16x16x32_f16(false, a0.v, false, f.b3, (short)0, acc[3], false, false);
    acc[4] = __builtin_amdgcn_wmma_f32_16x16x32_f16(false, a1.v, false, f.b0, (short)0, acc[4], false, false);
    acc[5] = __builtin_amdgcn_wmma_f32_16x16x32_f16(false, a1.v, false, f.b1, (short)0, acc[5], false, false);
    acc[6] = __builtin_amdgcn_wmma_f32_16x16x32_f16(false, a1.v, false, f.b2, (short)0, acc[6], false, false);
    acc[7] = __builtin_amdgcn_wmma_f32_16x16x32_f16(false, a1.v, false, f.b3, (short)0, acc[7], false, false);
  };

  // software pipeline: prologue load, rotate, peel last compute.
  // sched_barrier(0) fences keep "loads of step k+1" strictly above
  // "WMMAs of step k" so waits become partial instead of loadcnt==0.
  Frags cur = load_frags(0);
  SCHED_FENCE();
#pragma unroll
  for (int kk = 32; kk < CC; kk += 32) {
    Frags nxt = load_frags(kk);   // in flight while WMMAs below execute
    SCHED_FENCE();
    compute(cur);
    SCHED_FENCE();
    cur = nxt;
  }
  compute(cur);

  // C/D fp32 layout: VGPR r -> row (kSel*8 + r), column l16.
#pragma unroll
  for (int t = 0; t < 2; ++t) {
#pragma unroll
    for (int j = 0; j < 4; ++j) {
      const int n  = n0 + j * 16 + l16;
      const float bv = bias[n];
#pragma unroll
      for (int r = 0; r < 8; ++r) {
        const int m = m0 + t * 16 + kSel * 8 + r;
        Cout[(size_t)m * CC + n] = acc[t * 4 + j][r] + bv;
      }
    }
  }
}

// ---------------------------------------------------------------------------
// Neighborhood attention core (K=13 window, relative position bias, softmax).
// grid.x = B*L; block = 256 = 8 waves, wave h handles head h.
// Lane owns dims {2*lane, 2*lane+1} of DH=64. Output written as f16 (it is the
// A-matrix of the final projection GEMM).
// ---------------------------------------------------------------------------
__global__ __launch_bounds__(256) void natten1d(const float* __restrict__ qh,
                                                const float* __restrict__ kh,
                                                const float* __restrict__ vh,
                                                const float* __restrict__ rpb,
                                                _Float16* __restrict__ attn) {
  const int lane = threadIdx.x & 31;
  const int h    = threadIdx.x >> 5;
  const int bl   = blockIdx.x;            // b*L + l
  const int l    = bl & (LL - 1);
  const float SCALE = 0.125f;             // 64^-0.5

  int start = l - (KW / 2);
  if (start < 0) start = 0;
  if (start > LL - KW) start = LL - KW;

  const int d0 = lane * 2;
  const float* qp = qh + (size_t)bl * CC + h * DHD;
  const float q0 = qp[d0] * SCALE;
  const float q1 = qp[d0 + 1] * SCALE;

  // base row pointer for the window start: (b*L + start)*C + h*DH
  const size_t rowbase = ((size_t)(bl - l) + start) * CC + h * DHD;

  float logit[KW];
#pragma unroll
  for (int j = 0; j < KW; ++j) {
    const float* kp = kh + rowbase + (size_t)j * CC;
    float p = q0 * kp[d0] + q1 * kp[d0 + 1];
    p += __shfl_xor(p, 16, 32);
    p += __shfl_xor(p, 8, 32);
    p += __shfl_xor(p, 4, 32);
    p += __shfl_xor(p, 2, 32);
    p += __shfl_xor(p, 1, 32);
    logit[j] = p + rpb[h * (2 * KW - 1) + (start + j - l + KW - 1)];
  }

  float mx = logit[0];
#pragma unroll
  for (int j = 1; j < KW; ++j) mx = fmaxf(mx, logit[j]);
  float sum = 0.0f;
#pragma unroll
  for (int j = 0; j < KW; ++j) { logit[j] = expf(logit[j] - mx); sum += logit[j]; }
  const float inv = 1.0f / sum;

  float a0 = 0.0f, a1 = 0.0f;
#pragma unroll
  for (int j = 0; j < KW; ++j) {
    const float* vp = vh + rowbase + (size_t)j * CC;
    const float w = logit[j] * inv;
    a0 += w * vp[d0];
    a1 += w * vp[d0 + 1];
  }

  _Float16* op = attn + (size_t)bl * CC + h * DHD;
  op[d0]     = (_Float16)a0;
  op[d0 + 1] = (_Float16)a1;
}

// ---------------------------------------------------------------------------
extern "C" void kernel_launch(void* const* d_in, const int* in_sizes, int n_in,
                              void* d_out, int out_size, void* d_ws, size_t ws_size,
                              hipStream_t stream) {
  const float* q   = (const float*)d_in[0];
  const float* k   = (const float*)d_in[1];
  const float* v   = (const float*)d_in[2];
  const float* Wq  = (const float*)d_in[3];
  const float* bq  = (const float*)d_in[4];
  const float* Wk  = (const float*)d_in[5];
  const float* bk  = (const float*)d_in[6];
  const float* Wv  = (const float*)d_in[7];
  const float* bv  = (const float*)d_in[8];
  const float* rpb = (const float*)d_in[9];
  const float* Wo  = (const float*)d_in[10];
  const float* bo  = (const float*)d_in[11];
  float* out = (float*)d_out;

  const size_t NX = (size_t)BN * LL * CC;   // 4,194,304 elements
  const size_t NW = (size_t)CC * CC;        //   262,144 elements

  char* ws = (char*)d_ws;
  _Float16* Xq  = (_Float16*)ws; ws += NX * sizeof(_Float16);
  _Float16* Xk  = (_Float16*)ws; ws += NX * sizeof(_Float16);
  _Float16* Xv  = (_Float16*)ws; ws += NX * sizeof(_Float16);
  _Float16* WqT = (_Float16*)ws; ws += NW * sizeof(_Float16);
  _Float16* WkT = (_Float16*)ws; ws += NW * sizeof(_Float16);
  _Float16* WvT = (_Float16*)ws; ws += NW * sizeof(_Float16);
  _Float16* WoT = (_Float16*)ws; ws += NW * sizeof(_Float16);
  float*    qhb = (float*)ws;    ws += NX * sizeof(float);
  float*    khb = (float*)ws;    ws += NX * sizeof(float);
  float*    vhb = (float*)ws;    ws += NX * sizeof(float);
  _Float16* atn = (_Float16*)ws; ws += NX * sizeof(_Float16);

  const int cvtBlocks = (int)(NX / 256);
  cvt_f32_f16<<<cvtBlocks, 256, 0, stream>>>(q, Xq, (int)NX);
  cvt_f32_f16<<<cvtBlocks, 256, 0, stream>>>(k, Xk, (int)NX);
  cvt_f32_f16<<<cvtBlocks, 256, 0, stream>>>(v, Xv, (int)NX);

  const int wBlocks = (int)(NW / 256);
  cvt_transpose_f16<<<wBlocks, 256, 0, stream>>>(Wq, WqT);
  cvt_transpose_f16<<<wBlocks, 256, 0, stream>>>(Wk, WkT);
  cvt_transpose_f16<<<wBlocks, 256, 0, stream>>>(Wv, WvT);
  cvt_transpose_f16<<<wBlocks, 256, 0, stream>>>(Wo, WoT);

  const int gemmBlocks = (int)(BN * LL / 32);   // 256
  gemm_wmma_f16<<<gemmBlocks, 256, 0, stream>>>(Xq, WqT, bq, qhb);
  gemm_wmma_f16<<<gemmBlocks, 256, 0, stream>>>(Xk, WkT, bk, khb);
  gemm_wmma_f16<<<gemmBlocks, 256, 0, stream>>>(Xv, WvT, bv, vhb);

  natten1d<<<BN * LL, 256, 0, stream>>>(qhb, khb, vhb, rpb, atn);

  gemm_wmma_f16<<<gemmBlocks, 256, 0, stream>>>(atn, WoT, bo, out);
}
